// LinearAttention_80668075753539
// MI455X (gfx1250) — matrix operand order repaired and checked by reading, not compiled, not run on hardware
//
#include <hip/hip_runtime.h>
#include <hip/hip_bf16.h>

// ---------------------------------------------------------------------------
// GatedDeltaNet prefill for MI455X (gfx1250, wave32).
//   S=2048, HID=2048, H=16, DK=64, DV=128
// GEMMs via v_wmma_f32_16x16x32_bf16 (bf16 in, f32 accum), 32x64 wave tiles,
// ping-pong (copy-free) software-pipelined k-loop, launch_bounds to avoid
// accumulator spills, clamped (branch-free) boundary handling.
// Scan: state in registers, intra-wave shfl_xor reductions, no barriers.
// ---------------------------------------------------------------------------

#define SEQ   2048
#define HID   2048
#define NH    16
#define DK    64
#define DV    128
#define KEYD  (NH*DK)          // 1024
#define VALD  (NH*DV)          // 2048
#define OBIG  (2*KEYD + VALD + 2*NH)  // 2*1024 + 2048 + 2*16 = 4128

typedef __attribute__((ext_vector_type(16))) __bf16 v16bf;
typedef __attribute__((ext_vector_type(8)))  float  v8f;

// ---------------- converters ----------------
__global__ void cvt_bf16_kernel(const float* __restrict__ src,
                                __hip_bfloat16* __restrict__ dst, size_t n) {
    for (size_t i = (size_t)blockIdx.x * blockDim.x + threadIdx.x; i < n;
         i += (size_t)gridDim.x * blockDim.x)
        dst[i] = __float2bfloat16(src[i]);
}

// rows: [0,4096)=Wqkv, [4096,6144)=Wz, [6144,6160)=Wa, [6160,6176)=Wb
__global__ void pack_w_kernel(const float* __restrict__ qkv,
                              const float* __restrict__ z,
                              const float* __restrict__ a,
                              const float* __restrict__ b,
                              __hip_bfloat16* __restrict__ wb) {
    const size_t total = (size_t)OBIG * HID;
    for (size_t i = (size_t)blockIdx.x * blockDim.x + threadIdx.x; i < total;
         i += (size_t)gridDim.x * blockDim.x) {
        int r = (int)(i >> 11);        // / HID
        int c = (int)(i & (HID - 1));
        float v;
        if      (r < 4096) v = qkv[(size_t)r * HID + c];
        else if (r < 6144) v = z  [(size_t)(r - 4096) * HID + c];
        else if (r < 6160) v = a  [(size_t)(r - 6144) * HID + c];
        else               v = b  [(size_t)(r - 6160) * HID + c];
        wb[i] = __float2bfloat16(v);
    }
}

// ---------------- WMMA fragment loads --------------------------------------
// A frag (16x32 bf16): lane(m=l%16,h=l/16) holds A[m][k0+8h..+7], A[m][k0+16+8h..+7]
// B frag (32x16 bf16): lane(n=l%16,h=l/16) holds B[n][k0+8h..+7], B[n][k0+16+8h..+7]
__device__ __forceinline__ v16bf load_a_frag(const __hip_bfloat16* arow,
                                             int k0, int lhalf) {
    union { int4 q[2]; v16bf v; } u;
    u.q[0] = *(const int4*)(arow + k0 + 8 * lhalf);
    u.q[1] = *(const int4*)(arow + k0 + 16 + 8 * lhalf);
    return u.v;
}
__device__ __forceinline__ v16bf load_b_frag(const __hip_bfloat16* brow,
                                             int k0, int lhalf) {
    union { int4 q[2]; v16bf v; } u;
    const __hip_bfloat16* p = brow + k0 + 8 * lhalf;
    u.q[0] = *(const int4*)p;
    u.q[1] = *(const int4*)(p + 16);
    return u.v;
}

// ---------------- WMMA GEMM:  C[M,N] = A[M,K] * B[N,K]^T  (row-major) -------
// One wave computes a 32(M) x 64(N) tile: 2 A-frags x 4 B-frags = 8 WMMA/k-step.
// Boundary N tiles: clamp B row pointer (never stored), no EXEC divergence.
// k-loop: 2x-unrolled ping-pong buffers, peeled tail -> no copies, no branches
// in the steady state. Requires K % 64 == 0 (true for all uses here).
__global__ void __launch_bounds__(256, 1)
gemm_bf16_nt(const __hip_bfloat16* __restrict__ A,
             const __hip_bfloat16* __restrict__ B,
             float* __restrict__ C, int M, int N, int K) {
    const int tn    = (N + 63) >> 6;
    const int wid   = blockIdx.x * 8 + (threadIdx.x >> 5);
    const int total = (M >> 5) * tn;
    if (wid >= total) return;                    // wave-uniform exit (EXEC all-1 for WMMA)
    const int tm    = wid / tn;
    const int tc    = wid % tn;
    const int lane  = threadIdx.x & 31;
    const int lmod  = lane & 15;
    const int lhalf = lane >> 4;

    const __hip_bfloat16* arow0 = A + (size_t)(tm * 32 + lmod) * K;
    const __hip_bfloat16* arow1 = arow0 + (size_t)16 * K;

    const int nbase = tc * 64 + lmod;
    const __hip_bfloat16* brow[4];
    bool nvalid[4];
    #pragma unroll
    for (int j = 0; j < 4; ++j) {
        int n = nbase + 16 * j;
        nvalid[j] = (n < N);
        brow[j]   = B + (size_t)(nvalid[j] ? n : (N - 1)) * K;  // clamp, branch-free
    }

    v8f acc[2][4] = {};
    v16bf a0[2], b0[4], a1[2], b1[4];            // ping-pong fragment buffers

    // prologue: fill buffer 0 at k=0
    a0[0] = load_a_frag(arow0, 0, lhalf);
    a0[1] = load_a_frag(arow1, 0, lhalf);
    #pragma unroll
    for (int j = 0; j < 4; ++j) b0[j] = load_b_frag(brow[j], 0, lhalf);

    for (int k0 = 0; k0 < K - 64; k0 += 64) {
        // refill buffer 1 at k0+32 while consuming buffer 0
        a1[0] = load_a_frag(arow0, k0 + 32, lhalf);
        a1[1] = load_a_frag(arow1, k0 + 32, lhalf);
        #pragma unroll
        for (int j = 0; j < 4; ++j) b1[j] = load_b_frag(brow[j], k0 + 32, lhalf);
        #pragma unroll
        for (int i = 0; i < 2; ++i)
            #pragma unroll
            for (int j = 0; j < 4; ++j)
                acc[i][j] = __builtin_amdgcn_wmma_f32_16x16x32_bf16(
                    false, a0[i], false, b0[j], (short)0, acc[i][j], false, false);
        // refill buffer 0 at k0+64 while consuming buffer 1
        a0[0] = load_a_frag(arow0, k0 + 64, lhalf);
        a0[1] = load_a_frag(arow1, k0 + 64, lhalf);
        #pragma unroll
        for (int j = 0; j < 4; ++j) b0[j] = load_b_frag(brow[j], k0 + 64, lhalf);
        #pragma unroll
        for (int i = 0; i < 2; ++i)
            #pragma unroll
            for (int j = 0; j < 4; ++j)
                acc[i][j] = __builtin_amdgcn_wmma_f32_16x16x32_bf16(
                    false, a1[i], false, b1[j], (short)0, acc[i][j], false, false);
    }
    // peeled tail: k0 = K-64 (buffer 0 already holds K-64)
    {
        a1[0] = load_a_frag(arow0, K - 32, lhalf);
        a1[1] = load_a_frag(arow1, K - 32, lhalf);
        #pragma unroll
        for (int j = 0; j < 4; ++j) b1[j] = load_b_frag(brow[j], K - 32, lhalf);
        #pragma unroll
        for (int i = 0; i < 2; ++i)
            #pragma unroll
            for (int j = 0; j < 4; ++j)
                acc[i][j] = __builtin_amdgcn_wmma_f32_16x16x32_bf16(
                    false, a0[i], false, b0[j], (short)0, acc[i][j], false, false);
        #pragma unroll
        for (int i = 0; i < 2; ++i)
            #pragma unroll
            for (int j = 0; j < 4; ++j)
                acc[i][j] = __builtin_amdgcn_wmma_f32_16x16x32_bf16(
                    false, a1[i], false, b1[j], (short)0, acc[i][j], false, false);
    }
    // C frag: element r -> (m = r + 8*lhalf, n = lane%16)
    #pragma unroll
    for (int i = 0; i < 2; ++i) {
        #pragma unroll
        for (int j = 0; j < 4; ++j) {
            if (!nvalid[j]) continue;
            int n = nbase + 16 * j;
            #pragma unroll
            for (int r = 0; r < 8; ++r) {
                int mm = tm * 32 + i * 16 + r + 8 * lhalf;
                C[(size_t)mm * N + n] = acc[i][j][r];
            }
        }
    }
}

// ---------------- conv1d(K=4, causal, depthwise) + SiLU + l2norm + g/beta ---
__global__ void conv_kernel(const float* __restrict__ mixed,
                            const float* __restrict__ conv_w,
                            const float* __restrict__ A_log,
                            const float* __restrict__ dt_bias,
                            float* __restrict__ qb, float* __restrict__ kb,
                            float* __restrict__ vb, float* __restrict__ decayb,
                            float* __restrict__ betab) {
    __shared__ float act[2 * KEYD + VALD];   // 4096 floats
    __shared__ float scale[32];
    const int s = blockIdx.x;
    for (int c = threadIdx.x; c < 4096; c += blockDim.x) {
        const float* w = conv_w + c * 4;
        float acc = 0.f;
        #pragma unroll
        for (int i = 0; i < 4; ++i) {
            int ss = s - 3 + i;
            if (ss >= 0) acc += w[i] * mixed[(size_t)ss * OBIG + c];
        }
        acc = acc / (1.f + expf(-acc));      // SiLU
        act[c] = acc;
    }
    __syncthreads();
    if (threadIdx.x < 32) {                  // 16 heads x {q,k} l2-norm scales
        int hh = threadIdx.x & 15, which = threadIdx.x >> 4;
        const float* base = act + which * KEYD + hh * DK;
        float ss = 1e-6f;
        for (int d = 0; d < DK; ++d) ss += base[d] * base[d];
        float sc = rsqrtf(ss);
        if (which == 0) sc *= 0.125f;        // q also scaled by 1/sqrt(DK)
        scale[threadIdx.x] = sc;
    }
    if (threadIdx.x < NH) {                  // decay = exp(g), beta
        int hh = threadIdx.x;
        float araw = mixed[(size_t)s * OBIG + 2 * KEYD + VALD + hh] + dt_bias[hh];
        float sp   = (araw > 20.f) ? araw : log1pf(expf(araw));
        float g    = -expf(A_log[hh]) * sp;
        decayb[s * NH + hh] = expf(g);
        float braw = mixed[(size_t)s * OBIG + 2 * KEYD + VALD + NH + hh];
        betab[s * NH + hh] = 1.f / (1.f + expf(-braw));
    }
    __syncthreads();
    for (int c = threadIdx.x; c < 4096; c += blockDim.x) {
        if (c < KEYD) {
            qb[(size_t)s * KEYD + c] = act[c] * scale[c >> 6];
        } else if (c < 2 * KEYD) {
            int cc = c - KEYD;
            kb[(size_t)s * KEYD + cc] = act[c] * scale[16 + (cc >> 6)];
        } else {
            vb[(size_t)s * VALD + (c - 2 * KEYD)] = act[c];
        }
    }
}

// ---------------- delta-rule scan: barrier-free, state in registers --------
// grid = 16 (one block/head), block = 256 (8 waves).
// lane: v = wave*16 + (lane&15), khalf = lane>>4; 32 state regs per thread.
// Cross-lane reduction over k via single intra-wave shfl_xor(16) butterfly.
__global__ void scan_kernel(const float* __restrict__ qb,
                            const float* __restrict__ kb,
                            const float* __restrict__ vb,
                            const float* __restrict__ decayb,
                            const float* __restrict__ betab,
                            float* __restrict__ core) {
    const int h     = blockIdx.x;
    const int lane  = threadIdx.x & 31;
    const int wave  = threadIdx.x >> 5;
    const int v     = wave * 16 + (lane & 15);
    const int khalf = lane >> 4;
    float s[32];
    #pragma unroll
    for (int j = 0; j < 32; ++j) s[j] = 0.f;

    for (int t = 0; t < SEQ; ++t) {
        const size_t rb  = (size_t)t * NH + h;
        const float dec  = decayb[rb];
        const float bet  = betab[rb];
        const float vt   = vb[rb * DV + v];
        const float4* kp = (const float4*)(kb + rb * DK + khalf * 32);
        const float4* qp = (const float4*)(qb + rb * DK + khalf * 32);

        float4 kk[8];
        float kvp = 0.f;
        #pragma unroll
        for (int j = 0; j < 8; ++j) {        // S *= exp(g); kv = k . S
            kk[j] = kp[j];
            float4 sv = ((float4*)s)[j];
            sv.x *= dec; sv.y *= dec; sv.z *= dec; sv.w *= dec;
            kvp += kk[j].x * sv.x + kk[j].y * sv.y + kk[j].z * sv.z + kk[j].w * sv.w;
            ((float4*)s)[j] = sv;
        }
        kvp += __shfl_xor(kvp, 16);
        const float delta = (vt - kvp) * bet;

        float op = 0.f;
        #pragma unroll
        for (int j = 0; j < 8; ++j) {        // S += k (x) delta; o = q . S
            float4 qq = qp[j];
            float4 sv = ((float4*)s)[j];
            sv.x += kk[j].x * delta; sv.y += kk[j].y * delta;
            sv.z += kk[j].z * delta; sv.w += kk[j].w * delta;
            op += qq.x * sv.x + qq.y * sv.y + qq.z * sv.z + qq.w * sv.w;
            ((float4*)s)[j] = sv;
        }
        op += __shfl_xor(op, 16);
        if (khalf == 0) core[rb * DV + v] = op;
    }
}

// ---------------- RMSNorm over DV + SiLU(z) gate -> bf16 -------------------
// one wave per (s,h); 8 waves per block
__global__ void gate_kernel(const float* __restrict__ core,
                            const float* __restrict__ mixed,
                            const float* __restrict__ norm_w,
                            __hip_bfloat16* __restrict__ gated) {
    const int gw   = blockIdx.x * 8 + (threadIdx.x >> 5);   // gw = s*NH + h
    const int lane = threadIdx.x & 31;
    const int s    = gw >> 4;
    const int h    = gw & 15;
    const float* crow = core + (size_t)gw * DV;
    float c0[4], sq = 0.f;
    #pragma unroll
    for (int i = 0; i < 4; ++i) { c0[i] = crow[lane + 32 * i]; sq += c0[i] * c0[i]; }
    #pragma unroll
    for (int m = 16; m >= 1; m >>= 1) sq += __shfl_xor(sq, m);
    const float rs = rsqrtf(sq * (1.f / DV) + 1e-6f);
    const float* zrow = mixed + (size_t)s * OBIG + 2 * KEYD + h * DV;
    __hip_bfloat16* grow = gated + (size_t)s * VALD + h * DV;
    #pragma unroll
    for (int i = 0; i < 4; ++i) {
        int vv = lane + 32 * i;
        float zv = zrow[vv];
        float g  = c0[i] * rs * norm_w[vv] * (zv / (1.f + expf(-zv)));
        grow[vv] = __float2bfloat16(g);
    }
}

// ---------------- launcher -------------------------------------------------
extern "C" void kernel_launch(void* const* d_in, const int* in_sizes, int n_in,
                              void* d_out, int out_size, void* d_ws, size_t ws_size,
                              hipStream_t stream) {
    (void)in_sizes; (void)n_in; (void)out_size; (void)ws_size;
    const float* hs      = (const float*)d_in[0];
    const float* w_qkv   = (const float*)d_in[1];
    const float* w_a     = (const float*)d_in[2];
    const float* w_b     = (const float*)d_in[3];
    const float* w_z     = (const float*)d_in[4];
    const float* conv_w  = (const float*)d_in[5];
    const float* A_log   = (const float*)d_in[6];
    const float* dt_bias = (const float*)d_in[7];
    const float* norm_w  = (const float*)d_in[8];
    const float* out_p   = (const float*)d_in[9];
    float* out = (float*)d_out;

    char* ws = (char*)d_ws;
    // scratch layout (bytes)
    constexpr size_t SZ_HSB   = (size_t)SEQ * HID * 2;       // bf16
    constexpr size_t SZ_WB    = (size_t)OBIG * HID * 2;      // bf16
    constexpr size_t SZ_OUTW  = (size_t)HID * VALD * 2;      // bf16
    constexpr size_t SZ_MIXED = (size_t)SEQ * OBIG * 4;      // f32
    constexpr size_t SZ_QB    = (size_t)SEQ * KEYD * 4;
    constexpr size_t SZ_KB    = (size_t)SEQ * KEYD * 4;
    constexpr size_t SZ_VB    = (size_t)SEQ * VALD * 4;
    constexpr size_t SZ_DEC   = (size_t)SEQ * NH * 4;
    constexpr size_t SZ_BETA  = (size_t)SEQ * NH * 4;

    size_t off = 0;
    __hip_bfloat16* hsb   = (__hip_bfloat16*)(ws + off); off += SZ_HSB;
    __hip_bfloat16* wb    = (__hip_bfloat16*)(ws + off); off += SZ_WB;
    __hip_bfloat16* outw  = (__hip_bfloat16*)(ws + off); off += SZ_OUTW;
    float*          mixed = (float*)(ws + off);          off += SZ_MIXED;
    float*          qb    = (float*)(ws + off);          off += SZ_QB;
    float*          kb    = (float*)(ws + off);          off += SZ_KB;
    float*          vb    = (float*)(ws + off);          off += SZ_VB;
    float*          decb  = (float*)(ws + off);          off += SZ_DEC;
    float*          betab = (float*)(ws + off);          off += SZ_BETA;
    float*          core  = (float*)(ws + off);          off += (size_t)SEQ * VALD * 4;
    __hip_bfloat16* gated = hsb;   // hsb dead after GEMM1; same size as gated

    // 1) fp32 -> bf16 conversions + weight packing
    {
        size_t n = (size_t)SEQ * HID;
        cvt_bf16_kernel<<<(int)((n + 255) / 256), 256, 0, stream>>>(hs, hsb, n);
        pack_w_kernel<<<(int)(((size_t)OBIG * HID + 255) / 256), 256, 0, stream>>>(
            w_qkv, w_z, w_a, w_b, wb);
        size_t no = (size_t)HID * VALD;
        cvt_bf16_kernel<<<(int)((no + 255) / 256), 256, 0, stream>>>(out_p, outw, no);
    }
    // 2) mixed = hs @ [Wqkv;Wz;Wa;Wb]^T   (2048 x 4128)
    {
        int tn = (OBIG + 63) / 64;                 // 65
        int waves = (SEQ / 32) * tn;               // 4160
        gemm_bf16_nt<<<(waves + 7) / 8, 256, 0, stream>>>(hsb, wb, mixed,
                                                          SEQ, OBIG, HID);
    }
    // 3) conv + SiLU + l2norm + decay/beta
    conv_kernel<<<SEQ, 256, 0, stream>>>(mixed, conv_w, A_log, dt_bias,
                                         qb, kb, vb, decb, betab);
    // 4) delta-rule scan (one block per head)
    scan_kernel<<<NH, 256, 0, stream>>>(qb, kb, vb, decb, betab, core);
    // 5) RMSNorm + SiLU(z) gate -> bf16
    gate_kernel<<<SEQ * NH / 8, 256, 0, stream>>>(core, mixed, norm_w, gated);
    // 6) out = gated @ out_proj^T  (2048 x 2048)
    {
        int tn = (HID + 63) / 64;                  // 32
        int waves = (SEQ / 32) * tn;               // 2048
        gemm_bf16_nt<<<(waves + 7) / 8, 256, 0, stream>>>(gated, outw, out,
                                                          SEQ, HID, VALD);
    }
}
